// Generator_58128087384149
// MI455X (gfx1250) — compile-verified, matrix operand order checked by reading, not run on hardware
//
#include <hip/hip_runtime.h>

typedef __attribute__((ext_vector_type(16))) __bf16 v16bf;
typedef __attribute__((ext_vector_type(8)))  __bf16 v8bf;
typedef __attribute__((ext_vector_type(8)))  float  v8f;

#define Tt   2048
#define Nn   16
#define D192 192
#define NEGV (-1e16f)

__device__ __forceinline__ float leakyf(float x) { return x > 0.f ? x : 0.2f * x; }

// ---------------------------------------------------------------------------
// Prep: fold GCN matrices (M1 = Win^T@adj, M2 = Wout^T@adj) and W1h -> bf16
// ---------------------------------------------------------------------------
__global__ __launch_bounds__(256) void prep_kernel(
    const float* __restrict__ Wgin, const float* __restrict__ Wgout,
    const float* __restrict__ adj,  const float* __restrict__ W1h,
    float* __restrict__ ws_m, __bf16* __restrict__ w1hbf)
{
  const int bid = blockIdx.x, tid = threadIdx.x;
  if (bid == 0) {
    // M1[c][j] = sum_k Wgin[k][c]*adj[k][j]   (x @ Win^T @ adj == x @ M1)
    const int c = tid >> 4, j = tid & 15;
    float s1 = 0.f, s2 = 0.f;
#pragma unroll
    for (int k = 0; k < 16; ++k) {
      const float a = adj[k * 16 + j];
      s1 += Wgin [k * 16 + c] * a;
      s2 += Wgout[k * 16 + c] * a;
    }
    ws_m[tid]       = s1;
    ws_m[256 + tid] = s2;
  } else {
    const int idx = (bid - 1) * 256 + tid;      // 1536 blocks * 256 = 192*2048
    if (idx < D192 * Tt) w1hbf[idx] = (__bf16)W1h[idx];
  }
}

// ---------------------------------------------------------------------------
// 3-head attention: scores = (h@h^T)*graph, mask zeros, softmax, a@h+b, leaky
// Hm: [16][192] (per-head 64-wide slices), Sc: [3][16][16], Oc: [16][192]
// ---------------------------------------------------------------------------
__device__ void attn3(float* Hm, float* Sc, float* Oc,
                      const float* __restrict__ adj,
                      const float* __restrict__ bias, int tid)
{
  for (int e = tid; e < 768; e += 384) {
    const int head = e >> 8, r = e & 255, n = r >> 4, m = r & 15;
    const float* hn = Hm + n * D192 + head * 64;
    const float* hm = Hm + m * D192 + head * 64;
    float s = 0.f;
#pragma unroll 8
    for (int d = 0; d < 64; ++d) s += hn[d] * hm[d];
    s *= adj[n * 16 + m];
    Sc[e] = (s == 0.f) ? NEGV : s;
  }
  __syncthreads();
  if (tid < 48) {                                  // softmax per (head,row)
    float* row = Sc + tid * 16;
    float mx = row[0];
#pragma unroll
    for (int m = 1; m < 16; ++m) mx = fmaxf(mx, row[m]);
    float sum = 0.f;
#pragma unroll
    for (int m = 0; m < 16; ++m) { const float e = __expf(row[m] - mx); row[m] = e; sum += e; }
    const float inv = 1.f / sum;
#pragma unroll
    for (int m = 0; m < 16; ++m) row[m] *= inv;
  }
  __syncthreads();
  for (int e = tid; e < 3072; e += 384) {          // o = a@h + bias, leaky
    const int n = e / D192, col = e % D192, head = col >> 6;
    const float* a = Sc + head * 256 + n * 16;
    float s = 0.f;
#pragma unroll
    for (int m = 0; m < 16; ++m) s += a[m] * Hm[m * D192 + col];
    Oc[e] = leakyf(s + bias[col]);
  }
  __syncthreads();
}

// ---------------------------------------------------------------------------
// out-attention: h2 = Oc@W^T (16x16), single-head attention, +b, leaky -> Xout
// ---------------------------------------------------------------------------
__device__ void outatt(const float* Oc, const float* __restrict__ W,
                       const float* __restrict__ bias,
                       const float* __restrict__ adj,
                       float* h2, float* s2, float* Xout, int tid)
{
  if (tid < 256) {
    const int n = tid >> 4, j = tid & 15;
    float s = 0.f;
#pragma unroll 8
    for (int k = 0; k < D192; ++k) s += Oc[n * D192 + k] * W[j * D192 + k];
    h2[tid] = s;
  }
  __syncthreads();
  if (tid < 256) {
    const int n = tid >> 4, m = tid & 15;
    float s = 0.f;
#pragma unroll
    for (int j = 0; j < 16; ++j) s += h2[n * 16 + j] * h2[m * 16 + j];
    s *= adj[tid];
    s2[tid] = (s == 0.f) ? NEGV : s;
  }
  __syncthreads();
  if (tid < 16) {
    float* row = s2 + tid * 16;
    float mx = row[0];
#pragma unroll
    for (int m = 1; m < 16; ++m) mx = fmaxf(mx, row[m]);
    float sum = 0.f;
#pragma unroll
    for (int m = 0; m < 16; ++m) { const float e = __expf(row[m] - mx); row[m] = e; sum += e; }
    const float inv = 1.f / sum;
#pragma unroll
    for (int m = 0; m < 16; ++m) row[m] *= inv;
  }
  __syncthreads();
  if (tid < 256) {
    const int n = tid >> 4, j = tid & 15;
    float s = 0.f;
#pragma unroll
    for (int m = 0; m < 16; ++m) s += s2[n * 16 + m] * h2[m * 16 + j];
    Xout[tid] = leakyf(s + bias[j]);
  }
  __syncthreads();
}

// ---------------------------------------------------------------------------
// Main fused kernel: GCN -> LDS (bf16, transposed) -> WMMA GEMM -> GAT1 ->
// GAT2 -> FC.  One block per batch element; 12 waves (384 threads).
// ---------------------------------------------------------------------------
__global__ __launch_bounds__(384, 1) void gat_fused_kernel(
    const float* __restrict__ in_shots, const float* __restrict__ adj,
    const float* __restrict__ b1h, const float* __restrict__ W1o,
    const float* __restrict__ b1o, const float* __restrict__ W2h,
    const float* __restrict__ b2h, const float* __restrict__ W2o,
    const float* __restrict__ b2o, const float* __restrict__ Wfc,
    const float* __restrict__ bfc, const float* __restrict__ ws_m,
    const __bf16* __restrict__ w1hbf, float* __restrict__ out)
{
  __shared__ float smem[15360];                    // 60 KB
  __bf16* xs = (__bf16*)smem;                      // [16][1024] bf16 (32 KB)
  float*  Hb = smem + 8192;                        // [16][192] f32
  float*  Ms = smem + 8192;                        // overlay: M1[256] M2[256]
  float*  Ob = smem + 11264;                       // [16][192] f32
  float*  Sb = smem + 14336;                       // [3][16][16] scores (reused)
  float*  X1 = smem + 15104;                       // [16][16]

  const int tid  = threadIdx.x;
  const int lane = tid & 31;
  const int wave = tid >> 5;                       // 0..11
  const int b    = blockIdx.x;

  // preload folded GCN matrices into LDS (overlaid on Hb, dead until GEMM done)
  for (int i = tid; i < 512; i += 384) Ms[i] = ws_m[i];

  // ---- Phase 1: GCN staging + WMMA GEMM (K = 2048 in two 1024 chunks) ----
  v8f acc = {};
  const int row  = lane & 15;                      // A row (node n) / B col-in-tile
  const int half = lane >> 4;
  const int kswz = half * 8;                       // documented 16-bit K swizzle
  const int dcol = wave * 16 + row;                // 0..191

  for (int p = 0; p < 2; ++p) {
    __syncthreads();
    // stage GCN(in_shots[b]) transposed into LDS as bf16
    for (int t = tid; t < 1024; t += 384) {
      const int gt = p * 1024 + t;
      const float* xr = in_shots + ((size_t)b * Tt + gt) * Nn;
      float x0[16];
#pragma unroll
      for (int i = 0; i < 4; ++i) {
        const float4 v = ((const float4*)xr)[i];
        x0[4 * i] = v.x; x0[4 * i + 1] = v.y; x0[4 * i + 2] = v.z; x0[4 * i + 3] = v.w;
      }
      float v1[16];
#pragma unroll
      for (int j = 0; j < 4; ++j) {                // v1 = relu(x0 @ M1), 4-wide
        float4 s = {0.f, 0.f, 0.f, 0.f};
#pragma unroll
        for (int c = 0; c < 16; ++c) {
          const float4 m = ((const float4*)(Ms + c * 16))[j];
          s.x += x0[c] * m.x; s.y += x0[c] * m.y; s.z += x0[c] * m.z; s.w += x0[c] * m.w;
        }
        v1[4 * j]     = fmaxf(s.x, 0.f); v1[4 * j + 1] = fmaxf(s.y, 0.f);
        v1[4 * j + 2] = fmaxf(s.z, 0.f); v1[4 * j + 3] = fmaxf(s.w, 0.f);
      }
#pragma unroll
      for (int j = 0; j < 4; ++j) {                // v2 = v1 @ M2 -> LDS bf16
        float4 s = {0.f, 0.f, 0.f, 0.f};
#pragma unroll
        for (int c = 0; c < 16; ++c) {
          const float4 m = ((const float4*)(Ms + 256 + c * 16))[j];
          s.x += v1[c] * m.x; s.y += v1[c] * m.y; s.z += v1[c] * m.z; s.w += v1[c] * m.w;
        }
        xs[(4 * j)     * 1024 + t] = (__bf16)s.x;
        xs[(4 * j + 1) * 1024 + t] = (__bf16)s.y;
        xs[(4 * j + 2) * 1024 + t] = (__bf16)s.z;
        xs[(4 * j + 3) * 1024 + t] = (__bf16)s.w;
      }
    }
    __syncthreads();

    // WMMA: h[16, wave*16 .. wave*16+15] += X[16,1024] @ W1h^T chunk
    const __bf16* pa = xs + row * 1024 + kswz;
    const __bf16* pb = w1hbf + (size_t)dcol * Tt + p * 1024 + kswz;
#pragma unroll 4
    for (int k0 = 0; k0 < 1024; k0 += 32) {
      const v8bf alo = *(const v8bf*)(pa + k0);
      const v8bf ahi = *(const v8bf*)(pa + k0 + 16);
      const v8bf blo = *(const v8bf*)(pb + k0);
      const v8bf bhi = *(const v8bf*)(pb + k0 + 16);
      v16bf av, bv;
#pragma unroll
      for (int i = 0; i < 8; ++i) {
        av[i] = alo[i]; av[8 + i] = ahi[i];
        bv[i] = blo[i]; bv[8 + i] = bhi[i];
      }
      acc = __builtin_amdgcn_wmma_f32_16x16x32_bf16(
          false, av, false, bv, (short)0, acc, false, false);
    }
  }

  // store accumulator tile: D element (vgpr v, lane) -> row v+8*half, col lane&15
#pragma unroll
  for (int v = 0; v < 8; ++v)
    Hb[(v + half * 8) * D192 + wave * 16 + row] = acc[v];
  __syncthreads();

  // ---- GAT1 ----
  attn3(Hb, Sb, Ob, adj, b1h, tid);
  outatt(Ob, W1o, b1o, adj, Sb, Sb + 256, X1, tid);

  // ---- GAT2: h3 = x1 @ W2h^T (tiny) ----
  for (int e = tid; e < 3072; e += 384) {
    const int n = e / D192, col = e % D192;
    float s = 0.f;
#pragma unroll
    for (int m = 0; m < 16; ++m) s += X1[n * 16 + m] * W2h[col * 16 + m];
    Hb[e] = s;
  }
  __syncthreads();
  attn3(Hb, Sb, Ob, adj, b2h, tid);
  outatt(Ob, W2o, b2o, adj, Sb, Sb + 256, X1, tid);

  // ---- Final FC + transpose: out[b][k][n] ----
  if (tid < 48) {
    const int k = tid >> 4, n = tid & 15;
    float s = 0.f;
#pragma unroll
    for (int m = 0; m < 16; ++m) s += X1[n * 16 + m] * Wfc[k * 16 + m];
    out[(size_t)b * 48 + k * 16 + n] = s + bfc[k];
  }
}

extern "C" void kernel_launch(void* const* d_in, const int* in_sizes, int n_in,
                              void* d_out, int out_size, void* d_ws, size_t ws_size,
                              hipStream_t stream) {
  (void)in_sizes; (void)n_in; (void)out_size; (void)ws_size;
  const float* in_shots = (const float*)d_in[0];
  const float* adj      = (const float*)d_in[1];
  const float* Wgin     = (const float*)d_in[2];
  const float* Wgout    = (const float*)d_in[3];
  const float* W1h      = (const float*)d_in[4];
  const float* b1h      = (const float*)d_in[5];
  const float* W1o      = (const float*)d_in[6];
  const float* b1o      = (const float*)d_in[7];
  const float* W2h      = (const float*)d_in[8];
  const float* b2h      = (const float*)d_in[9];
  const float* W2o      = (const float*)d_in[10];
  const float* b2o      = (const float*)d_in[11];
  const float* Wfc      = (const float*)d_in[12];
  const float* bfc      = (const float*)d_in[13];
  float* out = (float*)d_out;

  float*  ws_m  = (float*)d_ws;                       // 512 floats: M1,M2
  __bf16* w1hbf = (__bf16*)((char*)d_ws + 2048);      // 192*2048 bf16

  prep_kernel<<<1537, 256, 0, stream>>>(Wgin, Wgout, adj, W1h, ws_m, w1hbf);
  gat_fused_kernel<<<4096, 384, 0, stream>>>(
      in_shots, adj, b1h, W1o, b1o, W2h, b2h, W2o, b2o, Wfc, bfc,
      ws_m, w1hbf, out);
}
